// Router_75436805587285
// MI455X (gfx1250) — matrix operand order repaired and checked by reading, not compiled, not run on hardware
//
#include <hip/hip_runtime.h>
#include <hip/hip_bf16.h>

#define TOKENS      32768
#define HIDDEN      2048
#define NUM_EXPERTS 64
#define TOP_K       2

// d_out flat layout (floats): weights [T*2] | indices [T*2] | scores [T*64] | aux [1]
#define OFF_W 0
#define OFF_I (TOKENS * TOP_K)
#define OFF_S (2 * TOKENS * TOP_K)
#define OFF_A (2 * TOKENS * TOP_K + TOKENS * NUM_EXPERTS)

// 64 * 0.01 / (32768^2 * 2)
#define AUX_SCALE 2.9802322387695312e-10f

typedef __attribute__((ext_vector_type(16))) __bf16 v16bf;
typedef __attribute__((ext_vector_type(8)))  float  v8f;

__global__ void router_zero_ws(float* ws) {
    ws[threadIdx.x] = 0.0f;   // 128 floats: [0..63]=agg_probs, [64..127]=tokens_per_expert
}

__global__ __launch_bounds__(256) void router_gemm_softmax_kernel(
    const float* __restrict__ x, const float* __restrict__ W,
    float* __restrict__ out, float* __restrict__ ws)
{
    __shared__ float s_logits[128][NUM_EXPERTS + 1];  // +1 pad vs bank conflicts
    __shared__ float s_agg[NUM_EXPERTS];
    __shared__ float s_cnt[NUM_EXPERTS];

    const int tid  = threadIdx.x;
    const int w    = tid >> 5;        // wave id 0..7
    const int lane = tid & 31;
    const int h    = lane >> 4;       // half-wave (K-split in WMMA layouts)
    const int nl   = lane & 15;       // M row (A) / N col (B)
    const int tok0 = blockIdx.x * 128 + w * 16;

    if (tid < NUM_EXPERTS) { s_agg[tid] = 0.0f; s_cnt[tid] = 0.0f; }

    const float* __restrict__ xrow = x + (size_t)(tok0 + nl) * HIDDEN;

    v8f acc[4];
    #pragma unroll
    for (int nt = 0; nt < 4; ++nt) acc[nt] = (v8f){0.f,0.f,0.f,0.f,0.f,0.f,0.f,0.f};

    // K loop: 64 iterations of v_wmma_f32_16x16x32_bf16 x 4 expert tiles.
    #pragma unroll 1
    for (int k0 = 0; k0 < HIDDEN; k0 += 32) {
        // Prefetch next A chunk of the streamed x row (L2 RT hint).
        __builtin_prefetch(xrow + k0 + 64, 0, 3);

        // ---- A fragment: 16x32 bf16. Lane (h,nl): K = k0 + h*16 + [0..15]
        const float4* pa0 = (const float4*)(xrow + k0 + h * 16);
        float4 a0 = pa0[0], a1 = pa0[1];
        float4 a2 = pa0[2], a3 = pa0[3];
        float af[16] = {a0.x,a0.y,a0.z,a0.w, a1.x,a1.y,a1.z,a1.w,
                        a2.x,a2.y,a2.z,a2.w, a3.x,a3.y,a3.z,a3.w};
        v16bf A;
        #pragma unroll
        for (int i = 0; i < 16; ++i) A[i] = (__bf16)af[i];

        // ---- B fragments: 32x16 bf16 per 16-expert tile. Lane (h,nl):
        // expert e = nt*16+nl, K = k0 + h*16 + [0..15]  (16 consecutive f32 of W row, L2-hot)
        #pragma unroll
        for (int nt = 0; nt < 4; ++nt) {
            const float4* pw = (const float4*)(W + (size_t)(nt * 16 + nl) * HIDDEN + k0 + h * 16);
            float4 b0 = pw[0], b1 = pw[1], b2 = pw[2], b3 = pw[3];
            float bf[16] = {b0.x,b0.y,b0.z,b0.w, b1.x,b1.y,b1.z,b1.w,
                            b2.x,b2.y,b2.z,b2.w, b3.x,b3.y,b3.z,b3.w};
            v16bf B;
            #pragma unroll
            for (int i = 0; i < 16; ++i) B[i] = (__bf16)bf[i];

            acc[nt] = __builtin_amdgcn_wmma_f32_16x16x32_bf16(
                false, A, false, B, (short)0, acc[nt], false, false);
        }
    }

    // ---- Spill logits tile to LDS in [token][expert] order.
    // C layout: vgpr v, lanes 0-15 -> M=v, lanes 16-31 -> M=v+8; N = nl.
    #pragma unroll
    for (int nt = 0; nt < 4; ++nt) {
        #pragma unroll
        for (int v = 0; v < 8; ++v) {
            s_logits[w * 16 + h * 8 + v][nt * 16 + nl] = acc[nt][v];
        }
    }
    __syncthreads();

    // ---- Phase 2: one thread per token -> softmax, top-2, scores, histogram.
    if (tid < 128) {
        const int t = blockIdx.x * 128 + tid;
        float* row = s_logits[tid];

        float mx = row[0];
        #pragma unroll
        for (int e = 1; e < NUM_EXPERTS; ++e) mx = fmaxf(mx, row[e]);

        float sum = 0.0f;
        #pragma unroll
        for (int e = 0; e < NUM_EXPERTS; ++e) {
            float v = __expf(row[e] - mx);
            row[e] = v;
            sum += v;
        }
        const float inv = 1.0f / sum;

        float m1 = -1.0f, m2 = -1.0f;
        int   i1 = 0,     i2 = 0;
        float* __restrict__ srow = out + OFF_S + (size_t)t * NUM_EXPERTS;
        #pragma unroll
        for (int e = 0; e < NUM_EXPERTS; ++e) {
            float p = row[e] * inv;
            srow[e] = p;
            atomicAdd(&s_agg[e], p);
            if (p > m1)      { m2 = m1; i2 = i1; m1 = p; i1 = e; }
            else if (p > m2) { m2 = p;  i2 = e; }
        }
        out[OFF_W + t * 2 + 0] = m1;
        out[OFF_W + t * 2 + 1] = m2;
        out[OFF_I + t * 2 + 0] = (float)i1;
        out[OFF_I + t * 2 + 1] = (float)i2;
        atomicAdd(&s_cnt[i1], 1.0f);
        atomicAdd(&s_cnt[i2], 1.0f);
    }
    __syncthreads();

    if (tid < NUM_EXPERTS) {
        atomicAdd(&ws[tid],               s_agg[tid]);
        atomicAdd(&ws[NUM_EXPERTS + tid], s_cnt[tid]);
    }
}

__global__ void router_aux_kernel(const float* __restrict__ ws, float* __restrict__ out) {
    __shared__ float red[NUM_EXPERTS];
    const int e = threadIdx.x;
    red[e] = ws[e] * ws[NUM_EXPERTS + e];
    __syncthreads();
    #pragma unroll
    for (int s = NUM_EXPERTS / 2; s > 0; s >>= 1) {
        if (e < s) red[e] += red[e + s];
        __syncthreads();
    }
    if (e == 0) out[OFF_A] = red[0] * AUX_SCALE;
}

extern "C" void kernel_launch(void* const* d_in, const int* in_sizes, int n_in,
                              void* d_out, int out_size, void* d_ws, size_t ws_size,
                              hipStream_t stream) {
    const float* x = (const float*)d_in[0];   // [32768, 2048] f32
    const float* W = (const float*)d_in[1];   // [64, 2048] f32
    float* out = (float*)d_out;
    float* ws  = (float*)d_ws;

    router_zero_ws<<<1, 128, 0, stream>>>(ws);
    router_gemm_softmax_kernel<<<TOKENS / 128, 256, 0, stream>>>(x, W, out, ws);
    router_aux_kernel<<<1, NUM_EXPERTS, 0, stream>>>(ws, out);
}
